// GraphConvolution_22308060135549
// MI455X (gfx1250) — matrix operand order repaired and checked by reading, not compiled, not run on hardware
//
#include <hip/hip_runtime.h>

#define BB 8
#define NN 2048
#define FF 128

typedef __attribute__((ext_vector_type(16))) _Float16 v16h;
typedef __attribute__((ext_vector_type(8)))  _Float16 v8h;
typedef __attribute__((ext_vector_type(4)))  _Float16 v4h;
typedef __attribute__((ext_vector_type(2)))  _Float16 v2h;
typedef __attribute__((ext_vector_type(8)))  float    v8f;

// ---- workgroup barrier that publishes ONLY LDS traffic ----
// __syncthreads() emits s_wait_loadcnt_dscnt 0x0 before the barrier, which
// forces our pipelined global loads to retire and kills latency hiding.
// We only need DScnt==0 (our ds_stores visible) before signal/wait.
__device__ __forceinline__ void wg_barrier_lds() {
  __atomic_signal_fence(__ATOMIC_SEQ_CST);   // compiler-only ordering
#if __has_builtin(__builtin_amdgcn_s_wait_dscnt) && \
    __has_builtin(__builtin_amdgcn_s_barrier_signal) && \
    __has_builtin(__builtin_amdgcn_s_barrier_wait)
  __builtin_amdgcn_s_wait_dscnt(0);
  __builtin_amdgcn_s_barrier_signal(-1);
  __builtin_amdgcn_s_barrier_wait(-1);
#else
  asm volatile("s_wait_dscnt 0x0\n\t"
               "s_barrier_signal -1\n\t"
               "s_barrier_wait -1" ::: "memory");
#endif
  __atomic_signal_fence(__ATOMIC_SEQ_CST);
}

// ---- fragment loaders (wave32, contiguous k order used by both A and B) ----
// A (16x32 f16): lane L<16 -> row M=L, K = kb..kb+15 contiguous
//                lanes 16..31: same rows, caller applies +16 to kb.
__device__ __forceinline__ v16h frag_a(const _Float16* row, int kb) {
  union { v16h v; v8h h[2]; } u;
  u.h[0] = *(const v8h*)(row + kb);
  u.h[1] = *(const v8h*)(row + kb + 8);
  return u.v;
}
// B (32x16 f16): lane L holds column N=L%16; K = kb..kb+15 contiguous
//                (caller applies +16 for lanes 16..31).
__device__ __forceinline__ v16h frag_b(const _Float16* row, int kb) {
  union { v16h v; v8h h[2]; } u;
  u.h[0] = *(const v8h*)(row + kb);
  u.h[1] = *(const v8h*)(row + kb + 8);
  return u.v;
}
__device__ __forceinline__ v8f wmma16(v16h a, v16h b, v8f c) {
  return __builtin_amdgcn_wmma_f32_16x16x32_f16(false, a, false, b, (short)0, c,
                                                false, false);
}

// ---------------- Pass 1: D^{-1/2} per row (one wave per row) ----------------
// Side effect: streams all of adj (134 MB) through L2 (192 MB), warming it
// for pass 2.
__global__ __launch_bounds__(256) void gcn_dinv(const float* __restrict__ adj,
                                                float* __restrict__ dinv) {
  const int lane = threadIdx.x & 31;
  const int wave = threadIdx.x >> 5;
  const int row  = blockIdx.x * 8 + wave;      // flat over B*N
  const float* r = adj + (size_t)row * NN;
  float s = 0.f;
  #pragma unroll 4
  for (int i = 0; i < NN / 128; ++i) {         // float4 per lane, 128 elems/iter
    float4 v = *(const float4*)(r + lane * 4 + i * 128);
    s += (v.x + v.y) + (v.z + v.w);
  }
  #pragma unroll
  for (int off = 16; off > 0; off >>= 1) s += __shfl_xor(s, off, 32);
  s += 1.0f;                                   // +I on the diagonal
  float di = (s > 0.f) ? rsqrtf(s) : 0.f;      // inf -> 0 for zero-degree rows
  if (lane == 0) dinv[row] = di;
}

// ---------------- Pass 2: fused  (d_inv (A+I) d_inv) @ x @ W^T + b ----------------
// Block: 256 threads (8 waves). Tile: 64 rows x 128 features of one batch.
// Register-pipelined: chunk k+1 global loads stay in flight across the
// LDS-only barrier and retire under chunk k's WMMA burst.
// LDS layout (dynamic, 62464 B total):
//   [0      , 9216 )  adjS : f16 [64][72]   support tile, col-scaled
//   [9216   , 27648)  xT   : f16 [128][72]  x^T chunk  (reused as hS f16 [64][136])
//   [27648  , 62464)  Wl   : f16 [128][136] W in native [o][f] layout (B-frag ready)
__global__ __launch_bounds__(256) void gcn_fused(const float* __restrict__ x,
                                                 const float* __restrict__ adj,
                                                 const float* __restrict__ W,
                                                 const float* __restrict__ bias,
                                                 const float* __restrict__ dinv,
                                                 float* __restrict__ out) {
  extern __shared__ char smem[];
  _Float16* adjS = (_Float16*)smem;                 // stride 72 (144B rows)
  _Float16* xT   = (_Float16*)(smem + 9216);        // stride 72
  _Float16* Wl   = (_Float16*)(smem + 27648);       // stride 136 (272B rows)
  _Float16* hS   = xT;                              // aliased after K-loop, stride 136

  const int b    = blockIdx.y;
  const int n0   = blockIdx.x * 64;
  const int t    = threadIdx.x;
  const int lane = t & 31;
  const int wave = t >> 5;
  const int wr   = wave >> 1;          // 4 row groups of 16
  const int wc   = wave & 1;           // 2 col groups of 64
  const int l16  = lane & 15;
  const int half = (lane >> 4) & 1;

  const float* adjB  = adj + (size_t)b * NN * NN;
  const float* xB    = x   + (size_t)b * NN * FF;
  const float* dinvB = dinv + (size_t)b * NN;

  // ---- stage W once, packed pairs: Wl[o][f] = W[o][f] (B-frag-ready layout)
  {
    const int f2 = (t & 63) * 2;
    const int o0 = t >> 6;             // 0..3
    #pragma unroll
    for (int i = 0; i < 32; ++i) {
      int o = o0 + i * 4;
      float2 w2 = *(const float2*)(W + o * FF + f2);
      v2h p = {(_Float16)w2.x, (_Float16)w2.y};
      *(v2h*)(Wl + o * 136 + f2) = p;
    }
  }

  v8f acc[4] = {};

  // staging index maps (fixed per thread)
  const int c4 = (t & 15) * 4;         // adj: 4 consecutive cols
  const int r0 = t >> 4;               // 0..15
  const int fX = t & 127;              // x: fixed feature
  const int m0 = (t >> 7) * 32;        // 0 or 32: contiguous m range

  // fragment base pointers (fixed per lane)
  const _Float16* arow  = adjS + (wr * 16 + l16) * 72;
  const _Float16* xrow0 = xT + (wc * 64 + 0 + l16) * 72;
  const _Float16* xrow1 = xT + (wc * 64 + 16 + l16) * 72;
  const _Float16* xrow2 = xT + (wc * 64 + 32 + l16) * 72;
  const _Float16* xrow3 = xT + (wc * 64 + 48 + l16) * 72;

  // ---- software pipeline registers: next chunk held in VGPRs
  float4 rA[4];
  float  rX0[16], rX1[16];

  // prologue: load chunk 0
  #pragma unroll
  for (int i = 0; i < 4; ++i)
    rA[i] = *(const float4*)(adjB + (size_t)(n0 + r0 + i * 16) * NN + c4);
  {
    const float* xp = xB + fX;
    #pragma unroll
    for (int i = 0; i < 16; ++i) {
      int m = m0 + i * 2;
      rX0[i] = xp[(size_t)m * FF];
      rX1[i] = xp[(size_t)(m + 1) * FF];
    }
  }

  for (int k0 = 0; k0 < NN; k0 += 64) {
    wg_barrier_lds();                            // prior fragment reads done
    // ---- convert + store current chunk (branchless diag, packed stores)
    {
      float4 dm4 = *(const float4*)(dinvB + k0 + c4);
      #pragma unroll
      for (int i = 0; i < 4; ++i) {
        int r = r0 + i * 16;
        float4 a = rA[i];
        int d = (n0 + r) - (k0 + c4);            // diagonal position in quad
        a.x += (d == 0) ? 1.f : 0.f;
        a.y += (d == 1) ? 1.f : 0.f;
        a.z += (d == 2) ? 1.f : 0.f;
        a.w += (d == 3) ? 1.f : 0.f;
        v4h p = {(_Float16)(a.x * dm4.x), (_Float16)(a.y * dm4.y),
                 (_Float16)(a.z * dm4.z), (_Float16)(a.w * dm4.w)};
        *(v4h*)(adjS + r * 72 + c4) = p;         // ds_store_b64
      }
      #pragma unroll
      for (int i = 0; i < 16; ++i) {
        int m = m0 + i * 2;
        v2h p = {(_Float16)rX0[i], (_Float16)rX1[i]};
        *(v2h*)(xT + fX * 72 + m) = p;           // ds_store_b32
      }
    }
    // ---- issue next chunk's global loads NOW; they stay in flight across
    //      the LDS-only barrier and retire under the WMMAs
    {
      const int kn = (k0 + 64 < NN) ? (k0 + 64) : 0;   // harmless wrap on last
      #pragma unroll
      for (int i = 0; i < 4; ++i) {
        const float* ap = adjB + (size_t)(n0 + r0 + i * 16) * NN + (kn + c4);
        rA[i] = *(const float4*)ap;
        __builtin_prefetch(ap + 64, 0, 0);       // speculative: chunk k+2
      }
      const float* xp = xB + (size_t)kn * FF + fX;
      #pragma unroll
      for (int i = 0; i < 16; ++i) {
        int m = m0 + i * 2;
        rX0[i] = xp[(size_t)m * FF];
        rX1[i] = xp[(size_t)(m + 1) * FF];
      }
    }
    wg_barrier_lds();

    // ---- 8x WMMA: batch all fragment loads per K-step, then burst the MMAs
    #pragma unroll
    for (int ks = 0; ks < 2; ++ks) {
      const int kbA = ks * 32 + half * 16;
      const int kbB = ks * 32 + half * 16;
      v16h af  = frag_a(arow, kbA);
      v16h bf0 = frag_b(xrow0, kbB);
      v16h bf1 = frag_b(xrow1, kbB);
      v16h bf2 = frag_b(xrow2, kbB);
      v16h bf3 = frag_b(xrow3, kbB);
      acc[0] = wmma16(af, bf0, acc[0]);
      acc[1] = wmma16(af, bf1, acc[1]);
      acc[2] = wmma16(af, bf2, acc[2]);
      acc[3] = wmma16(af, bf3, acc[3]);
    }
  }

  // ---- apply row scale d_inv[n]; spill h (f16) to LDS (aliases dead xT region)
  wg_barrier_lds();
  #pragma unroll
  for (int j = 0; j < 8; ++j) {
    int rloc = wr * 16 + half * 8 + j;           // C/D layout: M = j (+8 hi lanes)
    float dr = dinvB[n0 + rloc];
    #pragma unroll
    for (int tt = 0; tt < 4; ++tt) {
      hS[rloc * 136 + wc * 64 + tt * 16 + l16] = (_Float16)(acc[tt][j] * dr);
    }
  }
  wg_barrier_lds();

  // ---- GEMM2: out = h @ W^T + b   (K = F_IN = 128, four K=32 steps)
  v8f acc2[4] = {};
  const _Float16* hrow  = hS + (wr * 16 + l16) * 136;
  const _Float16* wrow0 = Wl + (wc * 64 + 0 + l16) * 136;
  const _Float16* wrow1 = Wl + (wc * 64 + 16 + l16) * 136;
  const _Float16* wrow2 = Wl + (wc * 64 + 32 + l16) * 136;
  const _Float16* wrow3 = Wl + (wc * 64 + 48 + l16) * 136;
  #pragma unroll
  for (int ks = 0; ks < 4; ++ks) {
    const int kbA = ks * 32 + half * 16;
    const int kbB = ks * 32 + half * 16;
    v16h af  = frag_a(hrow, kbA);
    v16h bf0 = frag_b(wrow0, kbB);
    v16h bf1 = frag_b(wrow1, kbB);
    v16h bf2 = frag_b(wrow2, kbB);
    v16h bf3 = frag_b(wrow3, kbB);
    acc2[0] = wmma16(af, bf0, acc2[0]);
    acc2[1] = wmma16(af, bf1, acc2[1]);
    acc2[2] = wmma16(af, bf2, acc2[2]);
    acc2[3] = wmma16(af, bf3, acc2[3]);
  }

  float* outB = out + (size_t)b * NN * FF;
  #pragma unroll
  for (int tt = 0; tt < 4; ++tt) {
    int o = wc * 64 + tt * 16 + l16;
    float bb = bias[o];
    #pragma unroll
    for (int j = 0; j < 8; ++j) {
      int n = n0 + wr * 16 + half * 8 + j;
      outB[(size_t)n * FF + o] = acc2[tt][j] + bb;
    }
  }
}

extern "C" void kernel_launch(void* const* d_in, const int* in_sizes, int n_in,
                              void* d_out, int out_size, void* d_ws, size_t ws_size,
                              hipStream_t stream) {
  (void)in_sizes; (void)n_in; (void)out_size; (void)ws_size;
  const float* x    = (const float*)d_in[0];   // [B,N,F_IN] f32
  const float* adj  = (const float*)d_in[1];   // [B,N,N]    f32
  const float* W    = (const float*)d_in[2];   // [F_OUT,F_IN] f32
  const float* bias = (const float*)d_in[3];   // [F_OUT]    f32
  float*       out  = (float*)d_out;           // [B,N,F_OUT] f32
  float*       dinv = (float*)d_ws;            // [B,N] f32 scratch (64 KB)

  gcn_dinv<<<dim3((BB * NN) / 8), dim3(256), 0, stream>>>(adj, dinv);

  dim3 grid(NN / 64, BB);
  size_t smem = 62464;  // adjS(9216) + xT(18432) + Wl(34816)
  gcn_fused<<<grid, dim3(256), smem, stream>>>(x, adj, W, bias, dinv, out);
}
